// HLGT_21225728376842
// MI455X (gfx1250) — matrix operand order repaired and checked by reading, not compiled, not run on hardware
//
#include <hip/hip_runtime.h>
#include <math.h>

#define NNODES  50000
#define NEDGES  800000
#define HIDDIM  128
#define NHEADS  2
#define HEADC   64
#define NLAYERS 6
#define FFDIM   256

typedef __attribute__((ext_vector_type(16))) _Float16 v16h;
typedef __attribute__((ext_vector_type(8)))  _Float16 v8h;
typedef __attribute__((ext_vector_type(8)))  float    v8f;

// ---------------- elementwise / setup kernels ----------------

__global__ void k_f32_to_f16(const float* __restrict__ a, _Float16* __restrict__ o, int n) {
  int i = blockIdx.x * blockDim.x + threadIdx.x;
  if (i < n) o[i] = (_Float16)a[i];
}

// Wt[n*K + k] = (f16) W[k*Nc + n]   (weights are [K x Nc] row-major in memory)
__global__ void k_transpose_f16(const float* __restrict__ W, _Float16* __restrict__ Wt,
                                int K, int Nc) {
  int i = blockIdx.x * blockDim.x + threadIdx.x;
  if (i >= K * Nc) return;
  int kk = i / Nc;
  int nn = i - kk * Nc;
  Wt[(size_t)nn * K + kk] = (_Float16)W[i];
}

__global__ void k_fill_u32(unsigned* __restrict__ p, unsigned v, int n) {
  int i = blockIdx.x * blockDim.x + threadIdx.x;
  if (i < n) p[i] = v;
}

__global__ void k_softmax_beta(const float* __restrict__ beta, float* __restrict__ betas, int n) {
  if (blockIdx.x == 0 && threadIdx.x == 0) {
    float m = -1e30f;
    for (int i = 0; i < n; ++i) m = fmaxf(m, beta[i]);
    float s = 0.f;
    for (int i = 0; i < n; ++i) { float e = expf(beta[i] - m); betas[i] = e; s += e; }
    float inv = 1.0f / s;
    for (int i = 0; i < n; ++i) betas[i] *= inv;
  }
}

__global__ void k_scale_store(float* __restrict__ out, const float* __restrict__ src,
                              const float* __restrict__ betas, int idx, int n) {
  int i = blockIdx.x * blockDim.x + threadIdx.x;
  if (i < n) out[i] = betas[idx] * src[i];
}

// ------- WMMA GEMM, 16x64 strip per wave: C[MxNc] = act(A[MxK](f16) @ W[KxNc] + bias) -------
// Bt is the transposed f16 weight, [Nc x K] row-major, so each output column's
// K-run is contiguous. One wave owns 4 adjacent 16x16 N-tiles: the A fragment is
// loaded once per K-chunk and reused by 4 WMMAs; the 4 B columns are reached via
// constant immediate offsets (j*16*K*2 bytes) off a single address register.
// ACT: 0 = none, 1 = relu, 2 = exact gelu.
template <int ACT>
__global__ void k_wmma_gemm(const _Float16* __restrict__ A, const _Float16* __restrict__ Bt,
                            const float* __restrict__ bias, float* __restrict__ C,
                            int M, int K, int Nc) {
  int wave = (int)((blockIdx.x * blockDim.x + threadIdx.x) >> 5);
  int lane = threadIdx.x & 31;
  int stripsN = Nc >> 6;                 // 64-wide output strips
  int strips  = (M >> 4) * stripsN;
  if (wave >= strips) return;            // wave-uniform: EXEC all-1s at WMMA
  int tM = (wave / stripsN) << 4;
  int tN = (wave % stripsN) << 6;
  int half = lane >> 4;                  // 0: lanes 0-15, 1: lanes 16-31
  int rl   = lane & 15;

  const _Float16* arow  = A  + (size_t)(tM + rl) * K;   // A row for this lane
  const _Float16* bbase = Bt + (size_t)(tN + rl) * K;   // first B column (contiguous K)

  v8f acc[4];
#pragma unroll
  for (int j = 0; j < 4; ++j) acc[j] = (v8f){0.f, 0.f, 0.f, 0.f, 0.f, 0.f, 0.f, 0.f};

  for (int kb = 0; kb < K; kb += 32) {
    // A fragment (16x32 f16): lanes<16 hold K {0..7,16..23}, lanes>=16 hold {8..15,24..31}
    v8h a0 = *(const v8h*)(arow + kb + 8 * half);
    v8h a1 = *(const v8h*)(arow + kb + 16 + 8 * half);
    v16h a;
#pragma unroll
    for (int i = 0; i < 8; ++i) { a[i] = a0[i]; a[i + 8] = a1[i]; }
#pragma unroll
    for (int j = 0; j < 4; ++j) {
      // B fragment (32x16 f16): lanes<16 hold K {0..7,16..23} of their column, lanes>=16 hold {8..15,24..31}
      const _Float16* bcol = bbase + (size_t)j * 16 * K;
      v8h b0 = *(const v8h*)(bcol + kb + 8 * half);
      v8h b1 = *(const v8h*)(bcol + kb + 16 + 8 * half);
      v16h b;
#pragma unroll
      for (int i = 0; i < 8; ++i) { b[i] = b0[i]; b[i + 8] = b1[i]; }
      acc[j] = __builtin_amdgcn_wmma_f32_16x16x32_f16(false, a, false, b, (short)0, acc[j],
                                                      false, false);
    }
  }

#pragma unroll
  for (int j = 0; j < 4; ++j) {
    int col  = tN + 16 * j + rl;
    float bv = bias ? bias[col] : 0.0f;
#pragma unroll
    for (int r = 0; r < 8; ++r) {
      float v = acc[j][r] + bv;
      if (ACT == 1)      v = fmaxf(v, 0.0f);
      else if (ACT == 2) v = 0.5f * v * (1.0f + erff(v * 0.70710678118654752f));
      C[(size_t)(tM + r + 8 * half) * Nc + col] = v;   // C/D layout: row = r + 8*half
    }
  }
}

// ---------------- edge attention kernels ----------------

// score[e*2+h] = dot(q[dst,h,:], k[src,h,:]) / 8 ; segment-max via ordered-uint atomicMax
__global__ void k_edge_score(const float* __restrict__ q, const float* __restrict__ k,
                             const int* __restrict__ ei, float* __restrict__ score,
                             unsigned* __restrict__ smaxk, int E) {
  int t = blockIdx.x * blockDim.x + threadIdx.x;
  if (t >= 2 * E) return;
  int e = t >> 1, h = t & 1;
  int src = ei[e], dst = ei[E + e];
  const float* qp = q + (size_t)dst * HIDDIM + h * HEADC;
  const float* kp = k + (size_t)src * HIDDIM + h * HEADC;
  float s = 0.f;
#pragma unroll 8
  for (int c = 0; c < HEADC; ++c) s += qp[c] * kp[c];
  s *= 0.125f;                           // 1/sqrt(64)
  score[t] = s;
  unsigned u = __float_as_uint(s);
  unsigned key = (u & 0x80000000u) ? ~u : (u | 0x80000000u);
  atomicMax(&smaxk[(size_t)dst * NHEADS + h], key);
}

// alpha = exp(score - smax[dst]); denom[dst] += alpha (alpha overwrites score)
__global__ void k_edge_alpha(float* __restrict__ score, const unsigned* __restrict__ smaxk,
                             float* __restrict__ denom, const int* __restrict__ ei, int E) {
  int t = blockIdx.x * blockDim.x + threadIdx.x;
  if (t >= 2 * E) return;
  int e = t >> 1, h = t & 1;
  int dst = ei[E + e];
  unsigned key = smaxk[(size_t)dst * NHEADS + h];
  unsigned u = (key & 0x80000000u) ? (key & 0x7FFFFFFFu) : ~key;
  float m = __uint_as_float(u);
  float a = expf(score[t] - m);
  score[t] = a;
  atomicAdd(&denom[(size_t)dst * NHEADS + h], a);
}

// out[dst,h,:] += (alpha/denom[dst,h]) * v[src,h,:]
__global__ void k_edge_scatter(const float* __restrict__ alpha, const float* __restrict__ denom,
                               const float* __restrict__ v, const int* __restrict__ ei,
                               float* __restrict__ out, int E) {
  int t = blockIdx.x * blockDim.x + threadIdx.x;
  if (t >= 2 * E) return;
  int e = t >> 1, h = t & 1;
  int src = ei[e], dst = ei[E + e];
  float w = alpha[t] / denom[(size_t)dst * NHEADS + h];
  const float* vp = v + (size_t)src * HIDDIM + h * HEADC;
  float* op = out + (size_t)dst * HIDDIM + h * HEADC;
#pragma unroll 4
  for (int c = 0; c < HEADC; ++c) atomicAdd(&op[c], w * vp[c]);
}

// ---------------- fused add + layernorm (+relu, +beta-accumulate), wave per node ----------------
__global__ void k_add_ln(const float* __restrict__ A, const float* __restrict__ B,
                         const float* __restrict__ g, const float* __restrict__ bt,
                         float* __restrict__ out, int n, int relu,
                         float* __restrict__ acc, const float* __restrict__ betas, int bidx) {
  int wid  = (int)((blockIdx.x * blockDim.x + threadIdx.x) >> 5);
  int lane = threadIdx.x & 31;
  if (wid >= n) return;
  size_t base = (size_t)wid * HIDDIM;
  float v[4];
  float s = 0.f;
#pragma unroll
  for (int j = 0; j < 4; ++j) {
    int c = lane + 32 * j;
    v[j] = A[base + c] + B[base + c];
    s += v[j];
  }
#pragma unroll
  for (int off = 16; off > 0; off >>= 1) s += __shfl_xor(s, off, 32);
  float mu = s * (1.0f / HIDDIM);
  float var = 0.f;
#pragma unroll
  for (int j = 0; j < 4; ++j) { float d = v[j] - mu; var += d * d; }
#pragma unroll
  for (int off = 16; off > 0; off >>= 1) var += __shfl_xor(var, off, 32);
  float inv = rsqrtf(var * (1.0f / HIDDIM) + 1e-5f);
  float scale = acc ? betas[bidx] : 0.f;
#pragma unroll
  for (int j = 0; j < 4; ++j) {
    int c = lane + 32 * j;
    float o = (v[j] - mu) * inv * g[c] + bt[c];
    if (relu) o = fmaxf(o, 0.0f);
    out[base + c] = o;
    if (acc) acc[base + c] += scale * o;
  }
}

// ---------------- launcher ----------------

static inline size_t alignup256(size_t x) { return (x + 255) & ~(size_t)255; }

extern "C" void kernel_launch(void* const* d_in, const int* in_sizes, int n_in,
                              void* d_out, int out_size, void* d_ws, size_t ws_size,
                              hipStream_t stream) {
  (void)in_sizes; (void)n_in; (void)out_size; (void)ws_size;
  const float* x    = (const float*)d_in[0];
  const int*   ei   = (const int*)d_in[1];     // [2,E] int32
  const float* Win  = (const float*)d_in[2];
  const float* b_in = (const float*)d_in[3];
  const float* Wq   = (const float*)d_in[4];
  const float* bq   = (const float*)d_in[5];
  const float* Wk   = (const float*)d_in[6];
  const float* bk   = (const float*)d_in[7];
  const float* Wv   = (const float*)d_in[8];
  const float* bv   = (const float*)d_in[9];
  const float* Wsk  = (const float*)d_in[10];
  const float* bsk  = (const float*)d_in[11];
  const float* W1   = (const float*)d_in[12];
  const float* b1   = (const float*)d_in[13];
  const float* W2   = (const float*)d_in[14];
  const float* b2   = (const float*)d_in[15];
  const float* g1   = (const float*)d_in[16];
  const float* bt1  = (const float*)d_in[17];
  const float* g2   = (const float*)d_in[18];
  const float* bt2  = (const float*)d_in[19];
  const float* beta = (const float*)d_in[20];

  const int NH = NNODES * HIDDIM;   // 6.4M
  const int NF = NNODES * FFDIM;    // 12.8M

  // ---- carve workspace ----
  char* w = (char*)d_ws;
  auto take = [&](size_t bytes) { void* p = (void*)w; w += alignup256(bytes); return p; };
  float*     z     = (float*)take((size_t)NH * 4);
  float*     q     = (float*)take((size_t)NH * 4);   // reused as FFN output
  float*     kbuf  = (float*)take((size_t)NH * 4);
  float*     vbuf  = (float*)take((size_t)NH * 4);
  float*     attn  = (float*)take((size_t)NH * 4);
  float*     h     = (float*)take((size_t)NH * 4);
  float*     ff    = (float*)take((size_t)NF * 4);
  _Float16*  z16   = (_Float16*)take((size_t)NH * 2);
  _Float16*  h16   = (_Float16*)take((size_t)NH * 2);
  _Float16*  ff16  = (_Float16*)take((size_t)NF * 2);
  float*     score = (float*)take((size_t)NEDGES * NHEADS * 4);
  unsigned*  smaxk = (unsigned*)take((size_t)NNODES * NHEADS * 4);
  float*     denom = (float*)take((size_t)NNODES * NHEADS * 4);
  _Float16*  wt    = (_Float16*)take(((size_t)HIDDIM * HIDDIM * (1 + 4 * NLAYERS)
                                      + (size_t)HIDDIM * FFDIM * 2 * NLAYERS) * 2);
  float*     betas = (float*)take(16 * 4);
  float*     acc   = (float*)d_out;

  const size_t HH = (size_t)HIDDIM * HIDDIM;
  const size_t HF = (size_t)HIDDIM * FFDIM;
  _Float16* Wint = wt;
  auto layer_wt = [&](int l) { return wt + HH + (size_t)l * (4 * HH + 2 * HF); };

  auto gemm = [&](int act, const _Float16* A, const _Float16* Bt, const float* bias,
                  float* C, int M, int K, int Nc) {
    int strips = (M >> 4) * (Nc >> 6);     // 16x64 strip per wave
    int blocks = (strips + 7) / 8;         // 8 waves (256 threads) per block
    if (act == 0)      k_wmma_gemm<0><<<blocks, 256, 0, stream>>>(A, Bt, bias, C, M, K, Nc);
    else if (act == 1) k_wmma_gemm<1><<<blocks, 256, 0, stream>>>(A, Bt, bias, C, M, K, Nc);
    else               k_wmma_gemm<2><<<blocks, 256, 0, stream>>>(A, Bt, bias, C, M, K, Nc);
  };
  auto cvt = [&](const float* a, _Float16* o, int n) {
    k_f32_to_f16<<<(n + 255) / 256, 256, 0, stream>>>(a, o, n);
  };
  auto trans = [&](const float* W, _Float16* Wt, int K, int Nc) {
    k_transpose_f16<<<(K * Nc + 255) / 256, 256, 0, stream>>>(W, Wt, K, Nc);
  };

  // ---- per-launch setup: betas + transposed f16 weights ----
  k_softmax_beta<<<1, 32, 0, stream>>>(beta, betas, NLAYERS + 1);
  trans(Win, Wint, HIDDIM, HIDDIM);
  for (int l = 0; l < NLAYERS; ++l) {
    _Float16* lw = layer_wt(l);
    trans(Wq  + (size_t)l * HH, lw,          HIDDIM, HIDDIM);
    trans(Wk  + (size_t)l * HH, lw + HH,     HIDDIM, HIDDIM);
    trans(Wv  + (size_t)l * HH, lw + 2 * HH, HIDDIM, HIDDIM);
    trans(Wsk + (size_t)l * HH, lw + 3 * HH, HIDDIM, HIDDIM);
    trans(W1  + (size_t)l * HF, lw + 4 * HH,      HIDDIM, FFDIM);   // [128x256] -> [256x128]
    trans(W2  + (size_t)l * HF, lw + 4 * HH + HF, FFDIM,  HIDDIM);  // [256x128] -> [128x256]
  }

  // ---- input projection: z = relu(x @ Win + b_in) ----
  cvt(x, z16, NH);
  gemm(1, z16, Wint, b_in, z, NNODES, HIDDIM, HIDDIM);
  k_scale_store<<<(NH + 255) / 256, 256, 0, stream>>>(acc, z, betas, 0, NH);
  cvt(z, z16, NH);

  const int egrid  = (2 * NEDGES + 255) / 256;
  const int ngrid  = (NNODES * NHEADS + 255) / 256;
  const int lngrid = (NNODES + 7) / 8;     // wave per node, 8 waves/block

  for (int l = 0; l < NLAYERS; ++l) {
    _Float16* lw = layer_wt(l);
    // Q/K/V and root skip (skip seeds attn buffer so the scatter adds on top)
    gemm(0, z16, lw,          bq  + (size_t)l * HIDDIM, q,    NNODES, HIDDIM, HIDDIM);
    gemm(0, z16, lw + HH,     bk  + (size_t)l * HIDDIM, kbuf, NNODES, HIDDIM, HIDDIM);
    gemm(0, z16, lw + 2 * HH, bv  + (size_t)l * HIDDIM, vbuf, NNODES, HIDDIM, HIDDIM);
    gemm(0, z16, lw + 3 * HH, bsk + (size_t)l * HIDDIM, attn, NNODES, HIDDIM, HIDDIM);
    // edge softmax attention
    k_fill_u32<<<ngrid, 256, 0, stream>>>(smaxk, 0u, NNODES * NHEADS);
    k_fill_u32<<<ngrid, 256, 0, stream>>>((unsigned*)denom, 0u, NNODES * NHEADS);
    k_edge_score  <<<egrid, 256, 0, stream>>>(q, kbuf, ei, score, smaxk, NEDGES);
    k_edge_alpha  <<<egrid, 256, 0, stream>>>(score, smaxk, denom, ei, NEDGES);
    k_edge_scatter<<<egrid, 256, 0, stream>>>(score, denom, vbuf, ei, attn, NEDGES);
    // h = LN(z + attn)
    k_add_ln<<<lngrid, 256, 0, stream>>>(z, attn, g1 + (size_t)l * HIDDIM,
                                         bt1 + (size_t)l * HIDDIM, h, NNODES, 0,
                                         nullptr, betas, 0);
    // FFN: ff = gelu(h @ W1 + b1); ffn = ff @ W2 + b2  (ffn reuses q)
    cvt(h, h16, NH);
    gemm(2, h16, lw + 4 * HH, b1 + (size_t)l * FFDIM, ff, NNODES, HIDDIM, FFDIM);
    cvt(ff, ff16, NF);
    gemm(0, ff16, lw + 4 * HH + HF, b2 + (size_t)l * HIDDIM, q, NNODES, FFDIM, HIDDIM);
    // z = relu(LN(h + ffn)); acc += betas[l+1] * z
    k_add_ln<<<lngrid, 256, 0, stream>>>(h, q, g2 + (size_t)l * HIDDIM,
                                         bt2 + (size_t)l * HIDDIM, z, NNODES, 1,
                                         acc, betas, l + 1);
    cvt(z, z16, NH);
  }
}
